// MOELayer_24309514895868
// MI455X (gfx1250) — compile-verified, matrix operand order checked by reading, not run on hardware
//
#include <hip/hip_runtime.h>
#include <math.h>

// ---------------------------------------------------------------------------
// MoE SwiGLU top-2, MI455X (gfx1250, wave32, WMMA bf16 16x16x32)
// Dims hardcoded to reference setup: B=4,S=2048,D=1024,F=4096,E=8,K=2
//
// - Weights pre-converted once into fragment-blocked bf16 (1KB chunks:
//   16 cols x 32 k; lane fragment = 32 contiguous bytes).
// - GEMM staging uses gfx1250 async global->LDS copies (ASYNCcnt) with
//   double-buffered LDS; fragments load as 2x ds_load_b128.
// - k-loop kept at unroll 1 so WMMA accumulates in place (no acc copies).
// ---------------------------------------------------------------------------

typedef __attribute__((ext_vector_type(16))) __bf16 v16bf;
typedef __attribute__((ext_vector_type(8)))  float  v8f;

constexpr int Ddim = 1024;
constexpr int Fdim = 4096;
constexpr int Edim = 8;
constexpr int KTOP = 2;
constexpr int T    = 4 * 2048;   // 8192 tokens

constexpr int BM = 64;  // rows per block tile
constexpr int BN = 64;  // cols per block tile

__device__ __forceinline__ unsigned short f2bf(float f) {
  unsigned int u = __builtin_bit_cast(unsigned int, f);
  unsigned int r = u + 0x7FFFu + ((u >> 16) & 1u);   // round-to-nearest-even
  return (unsigned short)(r >> 16);
}

// Low 32 bits of a flat LDS pointer are the DS-space byte offset.
__device__ __forceinline__ unsigned lds_addr(const void* p) {
  return (unsigned)(unsigned long long)p;
}
// gfx1250 async global->LDS 16-byte copy (tracked by ASYNCcnt).
__device__ __forceinline__ void async_copy_b128(unsigned lds, const void* g) {
  asm volatile("global_load_async_to_lds_b128 %0, %1, off"
               :: "v"(lds), "v"((unsigned long long)g) : "memory");
}
__device__ __forceinline__ void wait_async0() {
  asm volatile("s_wait_asynccnt 0x0" ::: "memory");
}

// Fragment-slot mapping (ISA 16-bit A/B layout, wave32):
//   element k (0..31) of a 16-wide fragment column n lives at
//   lane = (n&15) + 16*((k>>3)&1),  slot = 2*v + (k&1), v = (k>>4)*4 + ((k>>1)&3)

// ---------------------------------------------------------------------------
// Router: one wave per token. logits -> softmax -> top2; also emits bf16 x.
// ---------------------------------------------------------------------------
__global__ void __launch_bounds__(256)
router_kernel(const float* __restrict__ x, const float* __restrict__ Wg,
              const float* __restrict__ bg,
              unsigned short* __restrict__ xbf,  // [T][D] bf16
              float* __restrict__ probs,        // [T][E]
              int* __restrict__ counts,         // [E]
              int* __restrict__ tok_list,       // [E][T]
              float* __restrict__ gate_list,    // [E][T]
              int* __restrict__ slot_list)      // [E][T]
{
  int wave = threadIdx.x >> 5;
  int lane = threadIdx.x & 31;
  int t = blockIdx.x * 8 + wave;
  if (t >= T) return;

  const float* xr = x + (size_t)t * Ddim;
  float acc[Edim];
#pragma unroll
  for (int e = 0; e < Edim; ++e) acc[e] = 0.f;

  for (int d = lane; d < Ddim; d += 32) {
    float xv = xr[d];
    xbf[(size_t)t * Ddim + d] = f2bf(xv);
    const float4* wp = (const float4*)(Wg + (size_t)d * Edim);
    float4 w0 = wp[0], w1 = wp[1];
    acc[0] += xv * w0.x; acc[1] += xv * w0.y;
    acc[2] += xv * w0.z; acc[3] += xv * w0.w;
    acc[4] += xv * w1.x; acc[5] += xv * w1.y;
    acc[6] += xv * w1.z; acc[7] += xv * w1.w;
  }
#pragma unroll
  for (int e = 0; e < Edim; ++e) {
#pragma unroll
    for (int off = 16; off >= 1; off >>= 1)
      acc[e] += __shfl_xor(acc[e], off, 32);
  }

  if (lane == 0) {
    float lg[Edim], pr[Edim];
    float mx = -1e30f;
#pragma unroll
    for (int e = 0; e < Edim; ++e) { lg[e] = acc[e] + bg[e]; mx = fmaxf(mx, lg[e]); }
    float ssum = 0.f;
#pragma unroll
    for (int e = 0; e < Edim; ++e) { pr[e] = __expf(lg[e] - mx); ssum += pr[e]; }
    float inv = 1.f / ssum;
#pragma unroll
    for (int e = 0; e < Edim; ++e) {
      pr[e] *= inv;
      probs[(size_t)t * Edim + e] = pr[e];
    }
    // top-2, lowest index wins ties (matches jax.lax.top_k)
    int i0 = 0;
#pragma unroll
    for (int e = 1; e < Edim; ++e) if (pr[e] > pr[i0]) i0 = e;
    int i1 = -1; float v1 = -1.f;
#pragma unroll
    for (int e = 0; e < Edim; ++e) {
      if (e == i0) continue;
      if (pr[e] > v1) { v1 = pr[e]; i1 = e; }
    }
    int sel[2] = { i0, i1 };
#pragma unroll
    for (int k = 0; k < KTOP; ++k) {
      int e = sel[k];
      int pos = atomicAdd(&counts[e], 1);
      tok_list[e * T + pos]  = t;
      gate_list[e * T + pos] = pr[e];
      slot_list[e * T + pos] = k;
    }
  }
}

__global__ void init_kernel(int* counts) {
  if (threadIdx.x < Edim) counts[threadIdx.x] = 0;
}

__global__ void offsets_kernel(const int* __restrict__ counts, int* __restrict__ off) {
  if (threadIdx.x == 0) {
    int a = 0;
    for (int e = 0; e < Edim; ++e) { off[e] = a; a += counts[e]; }
  }
}

// Deterministic P_i sums (fixed-order strided + tree reduce)
__global__ void __launch_bounds__(256)
psum_kernel(const float* __restrict__ probs, float* __restrict__ Psum) {
  int e = blockIdx.x;
  __shared__ float sh[256];
  float s = 0.f;
  for (int t = threadIdx.x; t < T; t += 256) s += probs[(size_t)t * Edim + e];
  sh[threadIdx.x] = s;
  __syncthreads();
  for (int w = 128; w >= 1; w >>= 1) {
    if ((int)threadIdx.x < w) sh[threadIdx.x] += sh[threadIdx.x + w];
    __syncthreads();
  }
  if (threadIdx.x == 0) Psum[e] = sh[0];
}

// ---------------------------------------------------------------------------
// Weight pre-conversion: fp32 [K][N] row-major -> fragment-blocked bf16.
// ---------------------------------------------------------------------------
__global__ void __launch_bounds__(256)
wconv_kernel(const float* __restrict__ src, unsigned short* __restrict__ dst,
             int K, int N) {
  int e = blockIdx.y;
  const float* s = src + (size_t)e * K * N;
  unsigned short* d = dst + (size_t)e * K * N;
  int Ktiles = K >> 5;
  int ncols4 = N >> 2;
  int idx = blockIdx.x * 256 + threadIdx.x;
  int kp = idx / ncols4;
  if (kp >= (K >> 1)) return;
  int c4 = (idx - kp * ncols4) * 4;
  int k = kp * 2;

  const float4 r0 = *(const float4*)(s + (size_t)k * N + c4);
  const float4 r1 = *(const float4*)(s + (size_t)(k + 1) * N + c4);
  float a0[4] = { r0.x, r0.y, r0.z, r0.w };
  float a1[4] = { r1.x, r1.y, r1.z, r1.w };

  int kk = k & 31;
  int kt = k >> 5;
  int hi = (kk >> 3) & 1;
  int v  = ((kk >> 4) << 2) | ((kk >> 1) & 3);
#pragma unroll
  for (int j = 0; j < 4; ++j) {
    int col  = c4 + j;
    int nt   = col >> 4;
    int lane = (col & 15) + (hi << 4);
    unsigned pk = (unsigned)f2bf(a0[j]) | ((unsigned)f2bf(a1[j]) << 16);
    *(unsigned*)(d + ((size_t)(nt * Ktiles + kt) * 512) + lane * 16 + 2 * v) = pk;
  }
}

__device__ __forceinline__ v16bf load_frag(const unsigned short* F, int tile16, int lane) {
  return *(const v16bf*)(F + (tile16 * 32 + lane) * 16);
}

// ---------------------------------------------------------------------------
// Stage 1: per-expert grouped GEMM  h = silu(x@W1) * (x@W3)  -> bf16 hbuf
// Async double-buffered staging; 4 wmma per wave per k-step; unroll 1.
// ---------------------------------------------------------------------------
__global__ void __launch_bounds__(256)
moe_up_kernel(const unsigned short* __restrict__ xbf,
              const unsigned short* __restrict__ W1b,  // blocked bf16
              const unsigned short* __restrict__ W3b,
              const int* __restrict__ counts, const int* __restrict__ off,
              const int* __restrict__ tok_list,
              unsigned short* __restrict__ hbuf)   // [T*K][F] bf16
{
  int e   = blockIdx.z;
  int cnt = counts[e];
  int rowbase = blockIdx.y * BM;
  if (rowbase >= cnt) return;
  int n0 = blockIdx.x * BN;

  __shared__ int sTok[BM];
  __shared__ alignas(32) unsigned short AsF[2 * 2048];
  __shared__ alignas(32) unsigned short B1F[2 * 2048];
  __shared__ alignas(32) unsigned short B3F[2 * 2048];

  int tid  = threadIdx.x;
  int lane = tid & 31;
  int wave = tid >> 5;
  int wm = wave >> 1;  // 0..3 -> M sub-tile of 16
  int wn = wave & 1;   // 0..1 -> N sub-tile of 32

  if (tid < BM) {
    int i  = rowbase + tid;
    int ii = (i < cnt) ? i : (cnt - 1);
    sTok[tid] = tok_list[e * T + ii];
  }
  __syncthreads();

  // --- per-thread loop-invariant staging addresses ---
  int r = tid >> 2, s = tid & 3;
  int mtile = r >> 4, m = r & 15;
  int lane_d = m + ((s & 1) << 4), half = s >> 1;
  const unsigned short* aSrc = xbf + (size_t)sTok[r] * Ddim + s * 8;
  unsigned aL0 = lds_addr(&AsF[(mtile * 32 + lane_d) * 16 + half * 8]);

  constexpr int Ktiles = Ddim >> 5;   // 32
  const size_t cstride = (size_t)Ktiles * 512;
  int c = tid >> 6, s16 = (tid & 63) * 8;
  const unsigned short* b1Src =
      W1b + (size_t)e * Ddim * Fdim + ((size_t)(n0 >> 4) + c) * cstride + s16;
  const unsigned short* b3Src =
      W3b + (size_t)e * Ddim * Fdim + ((size_t)(n0 >> 4) + c) * cstride + s16;
  unsigned b1L0 = lds_addr(&B1F[c * 512 + s16]);
  unsigned b3L0 = lds_addr(&B3F[c * 512 + s16]);

  v8f acc1[2], acc3[2];
#pragma unroll
  for (int j = 0; j < 2; ++j)
#pragma unroll
    for (int i = 0; i < 8; ++i) { acc1[j][i] = 0.f; acc3[j][i] = 0.f; }

  // prologue: stage kt=0 into buffer 0
  async_copy_b128(aL0, aSrc);
  async_copy_b128(b1L0, b1Src);
  async_copy_b128(b3L0, b3Src);
  wait_async0();
  __syncthreads();

#pragma unroll 1
  for (int kt = 0; kt < Ktiles; ++kt) {
    unsigned curOff = (unsigned)(kt & 1) * 4096u;         // bytes
    if (kt + 1 < Ktiles) {   // stage next tile into the other buffer
      unsigned nxtOff = curOff ^ 4096u;
      async_copy_b128(aL0 + nxtOff, aSrc + ((kt + 1) << 5));
      async_copy_b128(b1L0 + nxtOff, b1Src + (size_t)(kt + 1) * 512);
      async_copy_b128(b3L0 + nxtOff, b3Src + (size_t)(kt + 1) * 512);
    }
    const unsigned short* Acur  = AsF + (curOff >> 1);
    const unsigned short* B1cur = B1F + (curOff >> 1);
    const unsigned short* B3cur = B3F + (curOff >> 1);

    v16bf afrag = load_frag(Acur, wm, lane);
#pragma unroll
    for (int j = 0; j < 2; ++j) {
      v16bf b1 = load_frag(B1cur, wn * 2 + j, lane);
      acc1[j] = __builtin_amdgcn_wmma_f32_16x16x32_bf16(
          false, afrag, false, b1, (short)0, acc1[j], false, false);
      v16bf b3 = load_frag(B3cur, wn * 2 + j, lane);
      acc3[j] = __builtin_amdgcn_wmma_f32_16x16x32_bf16(
          false, afrag, false, b3, (short)0, acc3[j], false, false);
    }
    wait_async0();
    __syncthreads();
  }

  // Epilogue: SwiGLU, write bf16 h rows (packed by expert offset)
  int offE  = off[e];
  int hi    = (lane >> 4) & 1;
  int col_l = lane & 15;
#pragma unroll
  for (int j = 0; j < 2; ++j) {
#pragma unroll
    for (int rr = 0; rr < 8; ++rr) {
      int row = wm * 16 + hi * 8 + rr;
      int gi  = rowbase + row;
      if (gi < cnt) {
        float h1 = acc1[j][rr], h3 = acc3[j][rr];
        float h  = (h1 / (1.f + __expf(-h1))) * h3;   // silu(h1)*h3
        hbuf[(size_t)(offE + gi) * Fdim + n0 + wn * 32 + j * 16 + col_l] = f2bf(h);
      }
    }
  }
}

// ---------------------------------------------------------------------------
// Stage 2: y = (h @ W2[e]) * gate  -> slotbuf[(2*token+slot)*D]
// ---------------------------------------------------------------------------
__global__ void __launch_bounds__(256)
moe_down_kernel(const unsigned short* __restrict__ hbuf,
                const unsigned short* __restrict__ W2b,  // blocked bf16
                const int* __restrict__ counts, const int* __restrict__ off,
                const int* __restrict__ tok_list,
                const float* __restrict__ gate_list,
                const int* __restrict__ slot_list,
                float* __restrict__ slotbuf)   // [T*K][D] fp32
{
  int e   = blockIdx.z;
  int cnt = counts[e];
  int rowbase = blockIdx.y * BM;
  if (rowbase >= cnt) return;
  int n0 = blockIdx.x * BN;

  __shared__ int   sRow[BM];
  __shared__ int   sDest[BM];
  __shared__ float sGate[BM];
  __shared__ alignas(32) unsigned short AsF[2 * 2048];
  __shared__ alignas(32) unsigned short B2F[2 * 2048];

  int tid  = threadIdx.x;
  int lane = tid & 31;
  int wave = tid >> 5;
  int wm = wave >> 1;
  int wn = wave & 1;
  int offE = off[e];

  if (tid < BM) {
    int i  = rowbase + tid;
    int ii = (i < cnt) ? i : (cnt - 1);
    sRow[tid]  = offE + ii;
    sDest[tid] = tok_list[e * T + ii] * KTOP + slot_list[e * T + ii];
    sGate[tid] = gate_list[e * T + ii];
  }
  __syncthreads();

  int r = tid >> 2, s = tid & 3;
  int mtile = r >> 4, m = r & 15;
  int lane_d = m + ((s & 1) << 4), half = s >> 1;
  const unsigned short* aSrc = hbuf + (size_t)sRow[r] * Fdim + s * 8;
  unsigned aL0 = lds_addr(&AsF[(mtile * 32 + lane_d) * 16 + half * 8]);

  constexpr int Ktiles = Fdim >> 5;   // 128
  const size_t cstride = (size_t)Ktiles * 512;
  int c = tid >> 6, s16 = (tid & 63) * 8;
  const unsigned short* b2Src =
      W2b + (size_t)e * Fdim * Ddim + ((size_t)(n0 >> 4) + c) * cstride + s16;
  unsigned b2L0 = lds_addr(&B2F[c * 512 + s16]);

  v8f acc[2];
#pragma unroll
  for (int j = 0; j < 2; ++j)
#pragma unroll
    for (int i = 0; i < 8; ++i) acc[j][i] = 0.f;

  async_copy_b128(aL0, aSrc);
  async_copy_b128(b2L0, b2Src);
  wait_async0();
  __syncthreads();

#pragma unroll 1
  for (int kt = 0; kt < Ktiles; ++kt) {
    unsigned curOff = (unsigned)(kt & 1) * 4096u;
    if (kt + 1 < Ktiles) {
      unsigned nxtOff = curOff ^ 4096u;
      async_copy_b128(aL0 + nxtOff, aSrc + ((kt + 1) << 5));
      async_copy_b128(b2L0 + nxtOff, b2Src + (size_t)(kt + 1) * 512);
    }
    const unsigned short* Acur  = AsF + (curOff >> 1);
    const unsigned short* B2cur = B2F + (curOff >> 1);

    v16bf afrag = load_frag(Acur, wm, lane);
#pragma unroll
    for (int j = 0; j < 2; ++j) {
      v16bf bf = load_frag(B2cur, wn * 2 + j, lane);
      acc[j] = __builtin_amdgcn_wmma_f32_16x16x32_bf16(
          false, afrag, false, bf, (short)0, acc[j], false, false);
    }
    wait_async0();
    __syncthreads();
  }

  int hi    = (lane >> 4) & 1;
  int col_l = lane & 15;
#pragma unroll
  for (int j = 0; j < 2; ++j) {
#pragma unroll
    for (int rr = 0; rr < 8; ++rr) {
      int row = wm * 16 + hi * 8 + rr;
      int gi  = rowbase + row;
      if (gi < cnt) {
        float y = acc[j][rr] * sGate[row];
        slotbuf[(size_t)sDest[row] * Ddim + n0 + wn * 32 + j * 16 + col_l] = y;
      }
    }
  }
}

// out[t][d] = slot0 + slot1 (fixed order -> deterministic)
__global__ void __launch_bounds__(256)
combine_kernel(const float* __restrict__ slotbuf, float* __restrict__ out) {
  size_t i = (size_t)blockIdx.x * 256 + threadIdx.x;
  if (i >= (size_t)T * Ddim) return;
  size_t t = i / Ddim, d = i - t * Ddim;
  out[i] = slotbuf[(t * 2) * Ddim + d] + slotbuf[(t * 2 + 1) * Ddim + d];
}

__global__ void aux_kernel(const int* __restrict__ counts,
                           const float* __restrict__ Psum,
                           float* __restrict__ out_aux) {
  if (threadIdx.x == 0 && blockIdx.x == 0) {
    float s = 0.f;
    for (int e = 0; e < Edim; ++e) {
      float f_i = (float)counts[e] / (float)(T * KTOP);
      float P_i = Psum[e] / (float)T;
      s += f_i * P_i;
    }
    out_aux[0] = (float)Edim * s;
  }
}

// ---------------------------------------------------------------------------
extern "C" void kernel_launch(void* const* d_in, const int* in_sizes, int n_in,
                              void* d_out, int out_size, void* d_ws, size_t ws_size,
                              hipStream_t stream) {
  const float* x  = (const float*)d_in[0];
  const float* Wg = (const float*)d_in[1];
  const float* bg = (const float*)d_in[2];
  const float* W1 = (const float*)d_in[3];
  const float* W3 = (const float*)d_in[4];
  const float* W2 = (const float*)d_in[5];
  float* out = (float*)d_out;

  // Workspace carving (~421 MB total)
  char* p = (char*)d_ws;
  auto carve = [&](size_t bytes) {
    char* r = p;
    p += (bytes + 255) & ~(size_t)255;
    return (void*)r;
  };
  unsigned short* xbf   = (unsigned short*)carve((size_t)T * Ddim * 2);
  unsigned short* hbuf  = (unsigned short*)carve((size_t)T * KTOP * Fdim * 2);
  float* slotbuf        = (float*)carve((size_t)T * KTOP * Ddim * 4);
  unsigned short* W1b   = (unsigned short*)carve((size_t)Edim * Ddim * Fdim * 2);
  unsigned short* W3b   = (unsigned short*)carve((size_t)Edim * Ddim * Fdim * 2);
  unsigned short* W2b   = (unsigned short*)carve((size_t)Edim * Fdim * Ddim * 2);
  float* probs          = (float*)carve((size_t)T * Edim * 4);
  int*   tok_list       = (int*)carve((size_t)Edim * T * 4);
  float* gate_list      = (float*)carve((size_t)Edim * T * 4);
  int*   slot_list      = (int*)carve((size_t)Edim * T * 4);
  int*   counts         = (int*)carve(256);
  int*   offs           = (int*)carve(256);
  float* Psum           = (float*)carve(256);

  init_kernel<<<1, 32, 0, stream>>>(counts);
  router_kernel<<<T / 8, 256, 0, stream>>>(x, Wg, bg, xbf, probs, counts,
                                           tok_list, gate_list, slot_list);
  offsets_kernel<<<1, 32, 0, stream>>>(counts, offs);
  psum_kernel<<<Edim, 256, 0, stream>>>(probs, Psum);

  // Pre-block weights to bf16 (once per call; memory-bound, ~600 MB total)
  {
    dim3 gup(((Ddim / 2) * (Fdim / 4)) / 256, Edim);   // (2048, 8)
    wconv_kernel<<<gup, 256, 0, stream>>>(W1, W1b, Ddim, Fdim);
    wconv_kernel<<<gup, 256, 0, stream>>>(W3, W3b, Ddim, Fdim);
    dim3 gdn(((Fdim / 2) * (Ddim / 4)) / 256, Edim);   // (2048, 8)
    wconv_kernel<<<gdn, 256, 0, stream>>>(W2, W2b, Fdim, Ddim);
  }

  dim3 g1(Fdim / BN, T / BM, Edim);   // (64,128,8); inactive tiles early-exit
  moe_up_kernel<<<g1, 256, 0, stream>>>(xbf, W1b, W3b, counts, offs, tok_list, hbuf);

  dim3 g2(Ddim / BN, T / BM, Edim);   // (16,128,8)
  moe_down_kernel<<<g2, 256, 0, stream>>>(hbuf, W2b, counts, offs, tok_list,
                                          gate_list, slot_list, slotbuf);

  combine_kernel<<<(int)(((size_t)T * Ddim + 255) / 256), 256, 0, stream>>>(slotbuf, out);
  aux_kernel<<<1, 32, 0, stream>>>(counts, Psum, out + (size_t)T * Ddim);
}
